// SoftGraphMatcher_40604620816920
// MI455X (gfx1250) — compile-verified, matrix operand order checked by reading, not run on hardware
//
#include <hip/hip_runtime.h>
#include <hip/hip_bf16.h>
#include <math.h>

typedef __attribute__((ext_vector_type(16))) _Float16 v16h;
typedef __attribute__((ext_vector_type(8)))  float    v8f;
typedef __attribute__((ext_vector_type(4)))  float    f4;

#define NROWS 8192
#define HDIM  256
#define EPSV  1e-8f

// ws layout (floats):
//   [0,256)        p1
//   [256,512)      p2
//   [512]          ||p1||,  [513] ||p2||
//   [1024, +8192)  m1
//   next 8192      m2
//   next 8192      obj  (wm[:,2], packed)
//   next 8192      subj (wm[:,0], packed)

__device__ __forceinline__ float sigmoidf_fast(float x) {
    return 1.0f / (1.0f + __expf(-x));
}

// --------------------------------------------------------------------------
// Kernel 1: p1 = pattern1 @ Wp.T + bp, p2 likewise; plus their L2 norms.
// One block of 256 threads (H = 256).
// --------------------------------------------------------------------------
__global__ void prep_patterns(const float* __restrict__ pat1,
                              const float* __restrict__ pat2,
                              const float* __restrict__ Wp,
                              const float* __restrict__ bp,
                              float* __restrict__ ws) {
    __shared__ float red1[256];
    __shared__ float red2[256];
    int h = threadIdx.x;

    float a0 = pat1[0], a1 = pat1[1], a2 = pat1[2];
    float b0 = pat2[0], b1 = pat2[1], b2 = pat2[2];
    float w0 = Wp[h * 3 + 0], w1 = Wp[h * 3 + 1], w2 = Wp[h * 3 + 2];
    float bh = bp[h];

    float p1h = fmaf(a0, w0, fmaf(a1, w1, fmaf(a2, w2, bh)));
    float p2h = fmaf(b0, w0, fmaf(b1, w1, fmaf(b2, w2, bh)));

    ws[h]        = p1h;
    ws[HDIM + h] = p2h;

    red1[h] = p1h * p1h;
    red2[h] = p2h * p2h;
    __syncthreads();
    for (int s = 128; s > 0; s >>= 1) {
        if (h < s) { red1[h] += red1[h + s]; red2[h] += red2[h + s]; }
        __syncthreads();
    }
    if (h == 0) {
        ws[2 * HDIM]     = sqrtf(red1[0]);
        ws[2 * HDIM + 1] = sqrtf(red2[0]);
    }
}

// --------------------------------------------------------------------------
// Kernel 2: per-row stats via WMMA.
// props = wm @ Wq.T + bq computed as 16x16 tiles with
// v_wmma_f32_16x16x32_f16 (K = 3 padded to 32).  Each wave owns one
// 16-row tile and loops over 16 column tiles, fusing +bq and accumulating
// ||row||^2, row.p1, row.p2, then lane-reduces and writes m1/m2/obj/subj.
// Block = 256 threads (8 waves), grid = 64 -> 512 row tiles.
// --------------------------------------------------------------------------
__global__ void row_stats_wmma(const float* __restrict__ wm,
                               const float* __restrict__ Wq,
                               const float* __restrict__ bq,
                               float* __restrict__ ws) {
    const float* p1 = ws;
    const float* p2 = ws + HDIM;
    const float  np1 = ws[2 * HDIM];
    const float  np2 = ws[2 * HDIM + 1];
    float* m1   = ws + 1024;
    float* m2   = m1 + NROWS;
    float* obj  = m2 + NROWS;
    float* subj = obj + NROWS;

    const int wave = threadIdx.x >> 5;
    const int lane = threadIdx.x & 31;
    const int tile = blockIdx.x * 8 + wave;   // 0..511
    const int row0 = tile * 16;

    // A: 16x32 f16 tile of wm rows; lanes 0-15 hold K=0..7 in halves 0..7
    // (we only need K=0..2), lanes 16-31 hold K=8..15 -> all zero.
    v16h a = {};
    if (lane < 16) {
        int r = row0 + lane;
        a[0] = (_Float16)wm[r * 3 + 0];
        a[1] = (_Float16)wm[r * 3 + 1];
        a[2] = (_Float16)wm[r * 3 + 2];
    }

    float n2[8], s1[8], s2[8];
#pragma unroll
    for (int v = 0; v < 8; ++v) { n2[v] = 0.f; s1[v] = 0.f; s2[v] = 0.f; }

    const int ncolLane = lane & 15;

#pragma unroll
    for (int ct = 0; ct < 16; ++ct) {
        // B: 32x16 f16, B[k][n] = Wq[ct*16+n, k]; lanes 0-15 hold K=0..7,16..23
        // for column n = lane (only K=0..2 nonzero), lanes 16-31 K=8..15,24..31 -> 0.
        v16h b = {};
        if (lane < 16) {
            int n = ct * 16 + lane;
            b[0] = (_Float16)Wq[n * 3 + 0];
            b[1] = (_Float16)Wq[n * 3 + 1];
            b[2] = (_Float16)Wq[n * 3 + 2];
        }
        v8f c = {};
        c = __builtin_amdgcn_wmma_f32_16x16x32_f16(
                /*neg_a=*/false, a, /*neg_b=*/false, b,
                /*c_mod=*/(short)0, c, /*reuse_a=*/false, /*reuse_b=*/false);

        // Every C element held by this lane has the same N = ct*16 + (lane&15).
        int col = ct * 16 + ncolLane;
        float bqv = bq[col];
        float p1v = p1[col];
        float p2v = p2[col];
#pragma unroll
        for (int v = 0; v < 8; ++v) {
            float t = c[v] + bqv;           // props[row, col]
            n2[v] = fmaf(t, t,   n2[v]);
            s1[v] = fmaf(t, p1v, s1[v]);
            s2[v] = fmaf(t, p2v, s2[v]);
        }
    }

    // Reduce over the 16 lanes of each half-wave (masks < 16 keep the two
    // halves independent).  Lane 0 ends with rows M=0..7, lane 16 with 8..15.
#pragma unroll
    for (int v = 0; v < 8; ++v) {
        for (int m = 1; m < 16; m <<= 1) {
            n2[v] += __shfl_xor(n2[v], m, 32);
            s1[v] += __shfl_xor(s1[v], m, 32);
            s2[v] += __shfl_xor(s2[v], m, 32);
        }
    }

    if ((lane & 15) == 0) {
        int mbase = row0 + ((lane >> 4) << 3);
#pragma unroll
        for (int v = 0; v < 8; ++v) {
            int r = mbase + v;
            float pn = sqrtf(n2[v]);
            float d1 = fmaxf(pn * np1, EPSV);
            float d2 = fmaxf(pn * np2, EPSV);
            m1[r]   = sigmoidf_fast(s1[v] / d1);
            m2[r]   = sigmoidf_fast(s2[v] / d2);
            obj[r]  = wm[r * 3 + 2];
            subj[r] = wm[r * 3 + 0];
        }
    }
}

// --------------------------------------------------------------------------
// Kernel 3: the bandwidth-bound N x N pass.
// out[i,j] = m1[i]*m2[j]*sigmoid(t / max(|t|, eps)), t = obj[i]*subj[j],
// diagonal zeroed.  One float4 NT store per thread (output 256MB > L2).
// Grid (NROWS/1024, NROWS), block 256.
// --------------------------------------------------------------------------
__device__ __forceinline__ float pair_val(float mi, float oi, float m2j, float sj) {
    float t = oi * sj;
    float d = fmaxf(fabsf(t), EPSV);
    float s = sigmoidf_fast(t / d);
    return mi * m2j * s;
}

__global__ void outer_kernel(const float* __restrict__ ws,
                             float* __restrict__ out) {
    const float* m1   = ws + 1024;
    const float* m2   = m1 + NROWS;
    const float* obj  = m2 + NROWS;
    const float* subj = obj + NROWS;

    const int i  = blockIdx.y;
    const int j0 = (blockIdx.x * blockDim.x + threadIdx.x) * 4;

    const float mi = m1[i];
    const float oi = obj[i];

    f4 m2v = *(const f4*)(m2 + j0);
    f4 sjv = *(const f4*)(subj + j0);

    f4 r;
    r.x = pair_val(mi, oi, m2v.x, sjv.x);
    r.y = pair_val(mi, oi, m2v.y, sjv.y);
    r.z = pair_val(mi, oi, m2v.z, sjv.z);
    r.w = pair_val(mi, oi, m2v.w, sjv.w);

    if (i - j0 >= 0 && i - j0 < 4) {
        int k = i - j0;
        if (k == 0) r.x = 0.f;
        else if (k == 1) r.y = 0.f;
        else if (k == 2) r.z = 0.f;
        else r.w = 0.f;
    }

    __builtin_nontemporal_store(r, (f4*)(out + (size_t)i * NROWS + j0));
}

// --------------------------------------------------------------------------
extern "C" void kernel_launch(void* const* d_in, const int* in_sizes, int n_in,
                              void* d_out, int out_size, void* d_ws, size_t ws_size,
                              hipStream_t stream) {
    const float* pattern1 = (const float*)d_in[0];
    const float* pattern2 = (const float*)d_in[1];
    const float* wm       = (const float*)d_in[2];
    const float* Wp       = (const float*)d_in[3];
    const float* bp       = (const float*)d_in[4];
    const float* Wq       = (const float*)d_in[5];
    const float* bq       = (const float*)d_in[6];
    float* ws  = (float*)d_ws;
    float* out = (float*)d_out;

    prep_patterns<<<1, 256, 0, stream>>>(pattern1, pattern2, Wp, bp, ws);
    row_stats_wmma<<<64, 256, 0, stream>>>(wm, Wq, bq, ws);
    dim3 g3(NROWS / 1024, NROWS);
    outer_kernel<<<g3, 256, 0, stream>>>(ws, out);
}
